// RVQ_25202868093348
// MI455X (gfx1250) — matrix operand order repaired and checked by reading, not compiled, not run on hardware
//
#include <hip/hip_runtime.h>

// ---------------- types ----------------
typedef __attribute__((ext_vector_type(4)))  float    vf4;
typedef __attribute__((ext_vector_type(8)))  float    v8f;
typedef __attribute__((ext_vector_type(8)))  __bf16   v8bf;
typedef __attribute__((ext_vector_type(16))) __bf16   v16bf;
typedef __attribute__((ext_vector_type(4)))  unsigned u32x4;
typedef __attribute__((ext_vector_type(4)))  int      i32x4;
typedef __attribute__((ext_vector_type(8)))  int      i32x8;
typedef __attribute__((ext_vector_type(4)))  unsigned uint4v;

// ---------------- problem constants ----------------
#define N_ROWS  65536
#define DDIM    512
#define VCODES  1024
#define KSTAGES 4

// ---------------- tiling constants ----------------
#define BLK_ROWS 64
#define THREADS  256                          // 8 waves: 4 row-tiles x 2 column groups
#define APITCH   1040                         // bytes per A row: 512 bf16 + 16B pad
#define BPITCH   1040                         // bytes per B-tile row (TDM pads to this)
#define BTILE_BYTES (16 * BPITCH)             // 16640
#define OFF_AHI   0                           // [64][520] bf16 hi residual
#define OFF_ALO   (64 * APITCH)               // 66560: [64][520] bf16 lo residual
#define OFF_B     (2 * 64 * APITCH)           // 133120: 8 slots (2 buf x 2 grp x hi/lo)
#define OFF_CODES (OFF_B + 8 * BTILE_BYTES)   // 266240: int[64][4]
#define OFF_VALS  (OFF_CODES + 64 * 4 * 4)    // 267264: float[2][64]
#define OFF_IDXS  (OFF_VALS + 2 * 64 * 4)     // 267776: int[2][64]
#define SMEM_BYTES (OFF_IDXS + 2 * 64 * 4)    // 268288 (< 320 KB)

// ---------------- WMMA ----------------
static __device__ __forceinline__ v8f wmma_bf16(v16bf a, v16bf b, v8f c) {
  return __builtin_amdgcn_wmma_f32_16x16x32_bf16(false, a, false, b, (short)0, c,
                                                 false, false);
}

// Load a 16-half fragment as two b128 chunks at byte distance `gap`.
static __device__ __forceinline__ v16bf ldfrag(const char* p, int gap) {
  v8bf lo = *(const v8bf*)p;
  v8bf hi = *(const v8bf*)(p + gap);
  v16bf r;
#pragma unroll
  for (int j = 0; j < 8; ++j) { r[j] = lo[j]; r[8 + j] = hi[j]; }
  return r;
}

// ---------------- Tensor Data Mover staging ----------------
#if __has_builtin(__builtin_amdgcn_tensor_load_to_lds)
#define RVQ_TDM 1
// One 16-row x 1024B tile (16 codewords x 512 bf16), LDS-padded 16B per 1024B row
// (pad_interval=256 dwords, pad_amount=4 dwords) -> 1040B LDS pitch, conflict-free b128.
static __device__ __forceinline__ void tdm_load_tile(const void* gsrc, unsigned ldsByteOff) {
  unsigned long long ga = (unsigned long long)(__SIZE_TYPE__)gsrc;
  u32x4 g0;
  g0[0] = 1u;                                  // count=1
  g0[1] = ldsByteOff;                          // lds_addr (dynamic LDS base assumed 0)
  g0[2] = (unsigned)ga;                        // global_addr[31:0]
  g0[3] = (unsigned)(ga >> 32) | (2u << 30);   // global_addr[56:32] | type=2
  i32x8 g1;
  g1[0] = (int)((2u << 16) | (1u << 20) | (7u << 22) | (3u << 25));
  g1[1] = (int)(256u << 16);                   // tensor_dim0 = 256 x 4B units
  g1[2] = (int)(4096u << 16);                  // tensor_dim1 = 4096 rows
  g1[3] = (int)(256u << 16);                   // tile_dim0 = 256 units
  g1[4] = 16;                                  // tile_dim1 = 16 rows
  g1[5] = 256;                                 // tensor_dim0_stride = 256 units
  g1[6] = 0;
  g1[7] = 0;
  i32x4 z4 = {0, 0, 0, 0};
#if __clang_major__ >= 23
  i32x8 z8 = {0, 0, 0, 0, 0, 0, 0, 0};
  __builtin_amdgcn_tensor_load_to_lds(g0, g1, z4, z4, z8, 0);
#else
  __builtin_amdgcn_tensor_load_to_lds(g0, g1, z4, z4, 0);
#endif
}
#endif

// Fallback staging: cooperative copy with the same padded layout.
static __device__ __forceinline__ void stage_tile(const __bf16* __restrict__ src,
                                                  char* __restrict__ dst) {
  for (int t = threadIdx.x; t < 1024; t += THREADS) {
    int row = t >> 6, ch = t & 63;
    *(uint4v*)(dst + row * BPITCH + ch * 16) =
        *(const uint4v*)((const char*)src + row * (DDIM * 2) + ch * 16);
  }
}

// ---------------- prep: split codebooks into bf16 hi/lo, 0.5*||c||^2 ----------------
__global__ __launch_bounds__(128)
void rvq_prep_kernel(const float* __restrict__ cb, __bf16* __restrict__ CBhi,
                     __bf16* __restrict__ CBlo, float* __restrict__ HC2) {
  const int v = blockIdx.x;                    // 0..4095
  const float* row = cb + (size_t)v * DDIM;
  float s = 0.f;
  for (int j = threadIdx.x; j < DDIM; j += 128) {
    float f = row[j];
    __bf16 h = (__bf16)f;
    CBhi[(size_t)v * DDIM + j] = h;
    CBlo[(size_t)v * DDIM + j] = (__bf16)(f - (float)h);
    s += f * f;
  }
#pragma unroll
  for (int m = 16; m > 0; m >>= 1) s += __shfl_xor(s, m, 32);
  __shared__ float red[4];
  if ((threadIdx.x & 31) == 0) red[threadIdx.x >> 5] = s;
  __syncthreads();
  if (threadIdx.x == 0) HC2[v] = 0.5f * (red[0] + red[1] + red[2] + red[3]);
}

// ---------------- fused RVQ encode ----------------
__global__ __launch_bounds__(THREADS, 1)
void rvq_main_kernel(const float* __restrict__ x, const float* __restrict__ cb,
                     const __bf16* __restrict__ CBhi, const __bf16* __restrict__ CBlo,
                     const float* __restrict__ HC2,
                     float* __restrict__ outCodes, float* __restrict__ outQ,
                     float* __restrict__ outR) {
  extern __shared__ char smem[];
  int*   codesAll = (int*)(smem + OFF_CODES);      // [64][4]
  float* valS     = (float*)(smem + OFF_VALS);     // [2][64]
  int*   idxS     = (int*)(smem + OFF_IDXS);       // [2][64]

  const int tid  = threadIdx.x;
  const int lane = tid & 31;
  const int wv   = tid >> 5;        // 8 waves
  const int r16  = lane & 15;
  const int hsel = lane >> 4;
  const int g    = wv >> 2;         // column group: 0 -> codes 0..511, 1 -> 512..1023
  const int wr   = wv & 3;          // row tile: rows wr*16 .. wr*16+15
  const int n0   = blockIdx.x * BLK_ROWS;

  // ---- init residual state = x, stored as (hi,lo) bf16 pair in A-frag layout ----
  for (int t = tid; t < BLK_ROWS * (DDIM / 4); t += THREADS) {
    int row = t >> 7, c4 = t & 127;
    vf4 xv = *(const vf4*)(x + (size_t)(n0 + row) * DDIM + c4 * 4);
    __bf16* ph = (__bf16*)(smem + OFF_AHI + row * APITCH + c4 * 8);
    __bf16* pl = (__bf16*)(smem + OFF_ALO + row * APITCH + c4 * 8);
#pragma unroll
    for (int j = 0; j < 4; ++j) {
      __bf16 h = (__bf16)xv[j];
      ph[j] = h;
      pl[j] = (__bf16)(xv[j] - (float)h);
    }
  }
  __syncthreads();

  const char* aH = smem + OFF_AHI + (wr * 16 + r16) * APITCH + hsel * 16;
  const char* aL = smem + OFF_ALO + (wr * 16 + r16) * APITCH + hsel * 16;
  const bool issuer = ((wv & 3) == 0);   // waves 0 and 4 drive staging for their group

  for (int k = 0; k < KSTAGES; ++k) {
    // prefetch first tile pair of this stage
#ifdef RVQ_TDM
    if (issuer) {
      unsigned slot = OFF_B + (unsigned)(g * 2) * BTILE_BYTES;   // buf 0
      int vbase = k * VCODES + g * 512;
      tdm_load_tile(CBhi + (size_t)vbase * DDIM, slot);
      tdm_load_tile(CBlo + (size_t)vbase * DDIM, slot + BTILE_BYTES);
    }
#endif
    float bv[8]; int bi[8];
#pragma unroll
    for (int i = 0; i < 8; ++i) { bv[i] = -__builtin_inff(); bi[i] = 0; }

    for (int ct = 0; ct < 32; ++ct) {
#ifdef RVQ_TDM
      if (issuer) __builtin_amdgcn_s_wait_tensorcnt(0);
      __syncthreads();                               // tile ct ready for everyone
      if (ct < 31 && issuer) {                       // prefetch ct+1 into other buffer
        unsigned slot = OFF_B + (unsigned)(((ct + 1) & 1) * 4 + g * 2) * BTILE_BYTES;
        int vbase = k * VCODES + g * 512 + (ct + 1) * 16;
        tdm_load_tile(CBhi + (size_t)vbase * DDIM, slot);
        tdm_load_tile(CBlo + (size_t)vbase * DDIM, slot + BTILE_BYTES);
      }
#else
      __syncthreads();                               // previous compute done
      {
        int vbase = k * VCODES + g * 512 + ct * 16;
        char* slot = smem + OFF_B + ((ct & 1) * 4 + g * 2) * BTILE_BYTES;
        stage_tile(CBhi + (size_t)vbase * DDIM, slot);
        stage_tile(CBlo + (size_t)vbase * DDIM, slot + BTILE_BYTES);
      }
      __syncthreads();
#endif
      // ---- 16x16 tile GEMM over D: pure ds_load + wmma ----
      const char* bB = smem + OFF_B + ((ct & 1) * 4 + g * 2) * BTILE_BYTES +
                       r16 * BPITCH + hsel * 16;
      v8f a0 = {}, a1 = {}, a2 = {};
#pragma unroll
      for (int kb = 0; kb < DDIM / 32; ++kb) {
        v16bf ah = ldfrag(aH + kb * 64, 32);
        v16bf al = ldfrag(aL + kb * 64, 32);
        v16bf bh = ldfrag(bB + kb * 64, 32);
        v16bf bl = ldfrag(bB + BTILE_BYTES + kb * 64, 32);
        a0 = wmma_bf16(ah, bh, a0);    // hi*hi
        a1 = wmma_bf16(ah, bl, a1);    // hi*lo
        a2 = wmma_bf16(al, bh, a2);    // lo*hi
      }
      const int colbase = g * 512 + ct * 16;
      float hc = HC2[k * VCODES + colbase + r16];
#pragma unroll
      for (int i = 0; i < 8; ++i) {
        float s = a0[i] + a1[i] + a2[i] - hc;
        if (s > bv[i]) { bv[i] = s; bi[i] = colbase + r16; }
      }
    }

    // ---- per-row argmax within each half-wave, then across column groups ----
#pragma unroll
    for (int i = 0; i < 8; ++i) {
      float v = bv[i]; int idx = bi[i];
#pragma unroll
      for (int m = 1; m <= 8; m <<= 1) {
        float ov = __shfl_xor(v, m, 32);
        int   oi = __shfl_xor(idx, m, 32);
        if (ov > v || (ov == v && oi < idx)) { v = ov; idx = oi; }
      }
      if (r16 == 0) {
        int row = wr * 16 + hsel * 8 + i;
        valS[g * 64 + row] = v;
        idxS[g * 64 + row] = idx;
      }
    }
    __syncthreads();
    if (tid < BLK_ROWS) {
      int row = tid;
      float v0 = valS[row], v1 = valS[64 + row];
      int code = (v1 > v0) ? idxS[64 + row] : idxS[row];   // ties -> lower (group 0)
      codesAll[row * KSTAGES + k] = code;
      outCodes[(size_t)(n0 + row) * KSTAGES + k] = (float)code;
    }
    __syncthreads();

    // ---- residual update: r = (hi+lo) - c, re-split (feeds scores only) ----
    if (k < KSTAGES - 1) {
      for (int t = tid; t < BLK_ROWS * (DDIM / 4); t += THREADS) {
        int row = t >> 7, c4 = t & 127;
        int code = codesAll[row * KSTAGES + k] & (VCODES - 1);
        vf4 cv = *(const vf4*)(cb + ((size_t)k * VCODES + code) * DDIM + c4 * 4);
        __bf16* ph = (__bf16*)(smem + OFF_AHI + row * APITCH + c4 * 8);
        __bf16* pl = (__bf16*)(smem + OFF_ALO + row * APITCH + c4 * 8);
#pragma unroll
        for (int j = 0; j < 4; ++j) {
          float r = (float)ph[j] + (float)pl[j] - cv[j];
          __bf16 h = (__bf16)r;
          ph[j] = h;
          pl[j] = (__bf16)(r - (float)h);
        }
      }
      __syncthreads();
    }
  }

  // ---- exact f32 epilogue replaying the reference op order ----
  __syncthreads();
  for (int t = tid; t < BLK_ROWS * (DDIM / 4); t += THREADS) {
    int row = t >> 7, c4 = t & 127;
    vf4 r = *(const vf4*)(x + (size_t)(n0 + row) * DDIM + c4 * 4);
    vf4 q = {0.f, 0.f, 0.f, 0.f};
#pragma unroll
    for (int k = 0; k < KSTAGES; ++k) {
      int code = codesAll[row * KSTAGES + k] & (VCODES - 1);
      vf4 cv = *(const vf4*)(cb + ((size_t)k * VCODES + code) * DDIM + c4 * 4);
      r = r - cv;       // residual: ((x - q0) - q1) - ...
      q = q + cv;       // quantized: 0 + q0 + q1 + ...
    }
    size_t o = (size_t)(n0 + row) * DDIM + c4 * 4;
    *(vf4*)(outQ + o) = q;
    *(vf4*)(outR + o) = r;
  }
}

// ---------------- host launch ----------------
extern "C" void kernel_launch(void* const* d_in, const int* in_sizes, int n_in,
                              void* d_out, int out_size, void* d_ws, size_t ws_size,
                              hipStream_t stream) {
  (void)in_sizes; (void)n_in; (void)out_size; (void)ws_size;
  const float* x  = (const float*)d_in[0];                 // [65536,512] f32
  const float* cb = (const float*)d_in[1];                 // [4,1024,512] f32

  __bf16* CBhi = (__bf16*)d_ws;                            // 4 MB
  __bf16* CBlo = CBhi + (size_t)KSTAGES * VCODES * DDIM;   // 4 MB
  float*  HC2  = (float*)(CBlo + (size_t)KSTAGES * VCODES * DDIM); // 16 KB

  float* outCodes = (float*)d_out;                         // N*K
  float* outQ = outCodes + (size_t)N_ROWS * KSTAGES;       // N*D
  float* outR = outQ + (size_t)N_ROWS * DDIM;              // N*D

  hipFuncSetAttribute(reinterpret_cast<const void*>(rvq_main_kernel),
                      hipFuncAttributeMaxDynamicSharedMemorySize, SMEM_BYTES);

  rvq_prep_kernel<<<KSTAGES * VCODES, 128, 0, stream>>>(cb, CBhi, CBlo, HC2);
  rvq_main_kernel<<<N_ROWS / BLK_ROWS, THREADS, SMEM_BYTES, stream>>>(
      x, cb, CBhi, CBlo, HC2, outCodes, outQ, outR);
}
